// LowDimQKMultiHeadAttention_25855703122319
// MI455X (gfx1250) — matrix operand order repaired and checked by reading, not compiled, not run on hardware
//
#include <hip/hip_runtime.h>
#include <stdint.h>

#define B_    2
#define TQ_   2048
#define TK_   2048
#define D_    1024
#define QK_   256
#define H_    8
#define HD_   32
#define OUT_  1024
#define HDV_  8192   /* H_ * D_ */

typedef __bf16 bf16;
typedef __attribute__((ext_vector_type(16))) __bf16 v16bf;
typedef __attribute__((ext_vector_type(8)))  __bf16 v8bf;
typedef __attribute__((ext_vector_type(8)))  float  v8f;
typedef __attribute__((ext_vector_type(4)))  float  v4f;
typedef __attribute__((ext_vector_type(4)))  int    v4i;

#if __has_builtin(__builtin_amdgcn_global_load_async_to_lds_b128) && \
    __has_builtin(__builtin_amdgcn_s_wait_asynccnt)
#define USE_ASYNC_LDS 1
#else
#define USE_ASYNC_LDS 0
#endif

// Native f32 -> bf16 (RNE on gfx1250; lowers to v_cvt_pk_bf16_f32 pairs)
static __device__ __forceinline__ bf16 f2bf(float x) { return (bf16)x; }

static __device__ __forceinline__ v16bf cat16(v8bf lo, v8bf hi) {
  v16bf r;
#pragma unroll
  for (int i = 0; i < 8; ++i) { r[i] = lo[i]; r[8 + i] = hi[i]; }
  return r;
}

static __device__ __forceinline__ v8f wmma_bf(v16bf a, v16bf b, v8f c) {
  // D = A(16x32 bf16) x B(32x16 bf16) + C(16x16 f32)
  return __builtin_amdgcn_wmma_f32_16x16x32_bf16(false, a, false, b, (short)0, c,
                                                 false, false);
}

// ---------------------------------------------------------------------------
// Kernel 1: convert Wo (OUT_ x HDV_ fp32) -> bf16 workspace copy
// ---------------------------------------------------------------------------
__global__ void k_cvt_wo(const float* __restrict__ wo, bf16* __restrict__ wob) {
  size_t i = ((size_t)blockIdx.x * blockDim.x + threadIdx.x) * 8;
  v4f v0 = *(const v4f*)(wo + i);
  v4f v1 = *(const v4f*)(wo + i + 4);
  v8bf r;
#pragma unroll
  for (int j = 0; j < 4; ++j) { r[j] = f2bf(v0[j]); r[4 + j] = f2bf(v1[j]); }
  *(v8bf*)(wob + i) = r;
}

// ---------------------------------------------------------------------------
// Kernel 2: V (B,TK,DV fp32) -> Vt (B,DV,TK bf16), 32x32 LDS tile transpose
// ---------------------------------------------------------------------------
__global__ void k_vt(const float* __restrict__ V, bf16* __restrict__ vt) {
  __shared__ float tile[32][33];
  int b  = blockIdx.z;
  int k0 = blockIdx.x * 32, d0 = blockIdx.y * 32;
  int tx = threadIdx.x & 31, ty = threadIdx.x >> 5;  // 32 x 8 threads
  const float* src = V + ((size_t)b * TK_ + k0) * D_ + d0;
#pragma unroll
  for (int i = 0; i < 32; i += 8) tile[ty + i][tx] = src[(size_t)(ty + i) * D_ + tx];
  __syncthreads();
  bf16* dst = vt + ((size_t)b * D_ + d0) * TK_ + k0;
#pragma unroll
  for (int i = 0; i < 32; i += 8) dst[(size_t)(ty + i) * TK_ + tx] = f2bf(tile[tx][ty + i]);
}

// ---------------------------------------------------------------------------
// Kernel 3: Q/K low-dim projections via WMMA.
//   Ql = (Q @ Wq^T + bq) * (1/sqrt(32))  -> (B*H, TQ, 32) bf16, row-major
//   Kl = (K @ Wk^T + bk)                 -> (B*H, TK, 32) bf16, row-major
// ---------------------------------------------------------------------------
__global__ void k_proj(const float* __restrict__ Q, const float* __restrict__ K,
                       const float* __restrict__ Wq, const float* __restrict__ bq,
                       const float* __restrict__ Wk, const float* __restrict__ bk,
                       bf16* __restrict__ ql, bf16* __restrict__ kl) {
  int wave = threadIdx.x >> 5, lane = threadIdx.x & 31;
  int idx = blockIdx.x * 8 + wave;
  int nt = idx & 15;  idx >>= 4;   // 16 N-tiles (QK_=256)
  int qt = idx & 127; idx >>= 7;   // 128 M-tiles (T=2048)
  int b  = idx & 1;   int sel = idx >> 1;  // 0 = Q, 1 = K

  const float* X    = sel ? K  : Q;
  const float* W    = sel ? Wk : Wq;
  const float* bias = sel ? bk : bq;
  bf16*        dst  = sel ? kl : ql;
  float outscale    = sel ? 1.0f : 0.17677669529663687f;  // 1/sqrt(32) folded into Ql

  bool hi = lane >= 16; int l16 = lane & 15;
  const float* arow = X + ((size_t)b * TQ_ + qt * 16 + l16) * D_;
  const float* brow = W + (size_t)(nt * 16 + l16) * D_;
  int ka0 = hi ? 8 : 0;    // A lane halves: K {0-7,16-23} / {8-15,24-31}
  int kb0 = hi ? 8 : 0;    // B lane halves: K {0-7,16-23} / {8-15,24-31}

  v8f acc = {0.f, 0.f, 0.f, 0.f, 0.f, 0.f, 0.f, 0.f};
  for (int k = 0; k < D_; k += 32) {
    v4f a0 = *(const v4f*)(arow + k + ka0);
    v4f a1 = *(const v4f*)(arow + k + ka0 + 4);
    v4f a2 = *(const v4f*)(arow + k + ka0 + 16);
    v4f a3 = *(const v4f*)(arow + k + ka0 + 20);
    v4f b0 = *(const v4f*)(brow + k + kb0);
    v4f b1 = *(const v4f*)(brow + k + kb0 + 4);
    v4f b2 = *(const v4f*)(brow + k + kb0 + 16);
    v4f b3 = *(const v4f*)(brow + k + kb0 + 20);
    v16bf av, bv;
#pragma unroll
    for (int j = 0; j < 4; ++j) {
      av[j]      = f2bf(a0[j]);  av[4 + j]  = f2bf(a1[j]);
      av[8 + j]  = f2bf(a2[j]);  av[12 + j] = f2bf(a3[j]);
      bv[j]      = f2bf(b0[j]);  bv[4 + j]  = f2bf(b1[j]);
      bv[8 + j]  = f2bf(b2[j]);  bv[12 + j] = f2bf(b3[j]);
    }
    acc = wmma_bf(av, bv, acc);
  }

  int n = nt * 16 + l16;
  int head = n >> 5, hd = n & 31;
  float bi = bias[n];
  size_t base = ((size_t)(b * H_ + head) * TQ_) * HD_ + hd;
  int m0 = qt * 16 + (hi ? 8 : 0);
#pragma unroll
  for (int i = 0; i < 8; ++i)
    dst[base + (size_t)(m0 + i) * HD_] = f2bf((acc[i] + bi) * outscale);
}

// ---------------------------------------------------------------------------
// Kernel 4: flash attention. Block = 8 waves = one (b,h,16-query) tile over
// full DV=1024 (wave w owns dv columns [w*128, w*128+128)).
// Kl tiles (32 keys x 32 hd bf16 = 2KB) are staged into LDS once per block via
// gfx1250 async global->LDS loads (double buffered, ASYNCcnt tracked); WMMA
// A-operands for the score matmul come from ds_load_b128.
// Scores computed transposed: St = Kl_tile(16x32) x Ql^T -> lane = query col.
// Online softmax lane-local + xor-16 shuffle. P^T built via one xor-16
// half-exchange, then O^T += V^T x P^T via WMMA.
// ---------------------------------------------------------------------------
__global__ void k_attn(const bf16* __restrict__ ql, const bf16* __restrict__ kl,
                       const bf16* __restrict__ vt, const uint8_t* __restrict__ mask,
                       bf16* __restrict__ ao) {
  __shared__ bf16 kbuf[2][32 * HD_];  // 2 x 2KB double buffer

  int wave = threadIdx.x >> 5, lane = threadIdx.x & 31;
  bool hi = lane >= 16; int l16 = lane & 15;
  int bh = blockIdx.y; int b = bh >> 3; int h = bh & 7;
  int q0 = blockIdx.x * 16;
  int ka0 = hi ? 8 : 0;   // A-operand K half offset, also mask VGPR->key offset

  // B operand: Ql^T (32x16). Lane = query column, K = hd dim.
  const bf16* qlrow = ql + ((size_t)bh * TQ_ + q0 + l16) * HD_ + (hi ? 8 : 0);
  v16bf bq_op = cat16(*(const v8bf*)qlrow, *(const v8bf*)(qlrow + 16));

  const bf16* klbase = kl + (size_t)bh * TK_ * HD_;
  const bf16* vtbase = vt + ((size_t)b * D_ + wave * 128) * TK_;
  const uint8_t* mrow = mask + (size_t)b * TK_;

  // stage one 32x32 bf16 Kl tile (contiguous 2KB) into kbuf[buf]
  auto stage = [&](int buf, int k0) {
    if (threadIdx.x < 128) {
      const bf16* g = klbase + (size_t)k0 * HD_ + threadIdx.x * 8;
      bf16* l = &kbuf[buf][threadIdx.x * 8];
#if USE_ASYNC_LDS
      __builtin_amdgcn_global_load_async_to_lds_b128(
          (__attribute__((address_space(1))) v4i*)g,
          (__attribute__((address_space(3))) v4i*)l, 0, 0);
#else
      *(v8bf*)l = *(const v8bf*)g;
#endif
    }
  };

  v8f acc[8];
#pragma unroll
  for (int t = 0; t < 8; ++t) acc[t] = (v8f){0.f, 0.f, 0.f, 0.f, 0.f, 0.f, 0.f, 0.f};
  float m_run = -__builtin_inff();
  float l_run = 0.f;

  stage(0, 0);  // prologue

  for (int k0 = 0; k0 < TK_; k0 += 32) {
    int cur = (k0 >> 5) & 1;
    bool more = (k0 + 32) < TK_;
    if (more) stage(cur ^ 1, k0 + 32);
#if USE_ASYNC_LDS
    if (more) __builtin_amdgcn_s_wait_asynccnt(1);   // tile `cur` complete
    else      __builtin_amdgcn_s_wait_asynccnt(0);
#endif
    __syncthreads();  // publish kbuf[cur] to all waves

    const bf16* kb = &kbuf[cur][0];
    v16bf aK1 = cat16(*(const v8bf*)(kb + l16 * HD_ + ka0),
                      *(const v8bf*)(kb + l16 * HD_ + ka0 + 16));
    v16bf aK2 = cat16(*(const v8bf*)(kb + (16 + l16) * HD_ + ka0),
                      *(const v8bf*)(kb + (16 + l16) * HD_ + ka0 + 16));
    if (more) __builtin_prefetch(vtbase + (size_t)l16 * TK_ + k0 + 32, 0, 0);

    v8f z = {0.f, 0.f, 0.f, 0.f, 0.f, 0.f, 0.f, 0.f};
    v8f s1 = wmma_bf(aK1, bq_op, z);   // St: keys k0+[0,16), query = l16
    v8f s2 = wmma_bf(aK2, bq_op, z);   // St: keys k0+[16,32)

    // key padding mask: VGPR i holds key k0 + ka0 + i (tile1), +16 (tile2)
    uint64_t m1 = *(const uint64_t*)(mrow + k0 + ka0);
    uint64_t m2 = *(const uint64_t*)(mrow + k0 + 16 + ka0);
#pragma unroll
    for (int i = 0; i < 8; ++i) {
      if ((m1 >> (8 * i)) & 0xffull) s1[i] = -__builtin_inff();
      if ((m2 >> (8 * i)) & 0xffull) s2[i] = -__builtin_inff();
    }

    // online softmax: lane-local 16 values + xor-16 partner reduction
    float lm = -__builtin_inff();
#pragma unroll
    for (int i = 0; i < 8; ++i) lm = fmaxf(lm, fmaxf(s1[i], s2[i]));
    float rm = fmaxf(lm, __shfl_xor(lm, 16, 32));
    float mnew = fmaxf(m_run, rm);
    float corr = (m_run < -1e30f) ? 0.f : __expf(m_run - mnew);
    bool dead = (mnew < -1e30f);

    float p1[8], p2[8], ls = 0.f;
#pragma unroll
    for (int i = 0; i < 8; ++i) {
      p1[i] = dead ? 0.f : __expf(s1[i] - mnew);
      p2[i] = dead ? 0.f : __expf(s2[i] - mnew);
      ls += p1[i] + p2[i];
    }
    float rs = ls + __shfl_xor(ls, 16, 32);
    l_run = l_run * corr + rs;
    m_run = mnew;

#pragma unroll
    for (int t = 0; t < 8; ++t)
#pragma unroll
      for (int i = 0; i < 8; ++i) acc[t][i] *= corr;

    // Build P^T (32x16 bf16) B-operand: lanes<16 need partner's tile1 (keys 8-15),
    // lanes>=16 need partner's tile2 (keys 16-23); one xor-16 exchange.
    v16bf bp;
#pragma unroll
    for (int i = 0; i < 8; ++i) {
      float send = hi ? p1[i] : p2[i];
      float recv = __shfl_xor(send, 16, 32);
      float e0 = hi ? recv : p1[i];   // K first-half element
      float e1 = hi ? p2[i] : recv;   // K second-half element
      bp[i]     = f2bf(e0);
      bp[8 + i] = f2bf(e1);
    }

    // O^T += V^T (16 dv x 32 keys) x P^T (32 keys x 16 queries)
#pragma unroll
    for (int t = 0; t < 8; ++t) {
      const bf16* vr = vtbase + (size_t)(t * 16 + l16) * TK_ + k0;
      v16bf aV = cat16(*(const v8bf*)(vr + 2 * ka0), *(const v8bf*)(vr + 2 * ka0 + 8));
      acc[t] = wmma_bf(aV, bp, acc[t]);
    }

    __syncthreads();  // release kbuf[cur] before it is overwritten next iter
  }

  float linv = (l_run > 0.f) ? 1.f / l_run : 0.f;
  // O^T tile: VGPR i = dv row (i or 8+i), lane column = query l16.
  bf16* aorow = ao + ((size_t)(b * TQ_ + q0 + l16)) * HDV_ + h * D_ + wave * 128 + (hi ? 8 : 0);
#pragma unroll
  for (int t = 0; t < 8; ++t) {
    v8bf pv;
#pragma unroll
    for (int i = 0; i < 8; ++i) pv[i] = f2bf(acc[t][i] * linv);
    *(v8bf*)(aorow + t * 16) = pv;
  }
}

// ---------------------------------------------------------------------------
// Kernel 5: output projection. (B*TQ, 8192) bf16 @ Wo^T(8192,1024) + bo -> f32
// One wave = one 16x16 tile, K = 8192 in 32-steps.
// ---------------------------------------------------------------------------
__global__ void k_oproj(const bf16* __restrict__ ao, const bf16* __restrict__ wob,
                        const float* __restrict__ bo, float* __restrict__ out) {
  int wave = threadIdx.x >> 5, lane = threadIdx.x & 31;
  bool hi = lane >= 16; int l16 = lane & 15;
  int idx = blockIdx.x * 8 + wave;
  int nt = idx & 63;       // 64 N-tiles (OUT_=1024)
  int mt = idx >> 6;       // 256 M-tiles (B*TQ=4096)
  const bf16* arow = ao  + (size_t)(mt * 16 + l16) * HDV_;
  const bf16* brow = wob + (size_t)(nt * 16 + l16) * HDV_ + (hi ? 16 : 0);
  int ka0 = hi ? 8 : 0;

  v8f acc = {0.f, 0.f, 0.f, 0.f, 0.f, 0.f, 0.f, 0.f};
  for (int k = 0; k < HDV_; k += 32) {
    v16bf av = cat16(*(const v8bf*)(arow + k + 2 * ka0), *(const v8bf*)(arow + k + 2 * ka0 + 8));
    v16bf bv = cat16(*(const v8bf*)(brow + k), *(const v8bf*)(brow + k + 8));
    acc = wmma_bf(av, bv, acc);
  }
  float bi = bo[nt * 16 + l16];
  int m0 = mt * 16 + (hi ? 8 : 0);
#pragma unroll
  for (int i = 0; i < 8; ++i)
    out[(size_t)(m0 + i) * OUT_ + nt * 16 + l16] = acc[i] + bi;
}

// ---------------------------------------------------------------------------
extern "C" void kernel_launch(void* const* d_in, const int* in_sizes, int n_in,
                              void* d_out, int out_size, void* d_ws, size_t ws_size,
                              hipStream_t stream) {
  (void)in_sizes; (void)n_in; (void)out_size; (void)ws_size;
  const float*   Q    = (const float*)d_in[0];
  const float*   K    = (const float*)d_in[1];
  const float*   V    = (const float*)d_in[2];
  const float*   Wq   = (const float*)d_in[3];
  const float*   bq   = (const float*)d_in[4];
  const float*   Wk   = (const float*)d_in[5];
  const float*   bk   = (const float*)d_in[6];
  const float*   Wo   = (const float*)d_in[7];
  const float*   bo   = (const float*)d_in[8];
  const uint8_t* mask = (const uint8_t*)d_in[9];
  float* out = (float*)d_out;

  // Workspace layout (bf16 elements)
  bf16* ql  = (bf16*)d_ws;                               // B*H*TQ*HD
  bf16* kl  = ql  + (size_t)B_ * H_ * TQ_ * HD_;         // B*H*TK*HD
  bf16* vt  = kl  + (size_t)B_ * H_ * TK_ * HD_;         // B*DV*TK
  bf16* wob = vt  + (size_t)B_ * D_ * TK_;               // OUT*HDV
  bf16* ao  = wob + (size_t)OUT_ * HDV_;                 // B*TQ*HDV

  // 1) Wo -> bf16
  k_cvt_wo<<<(OUT_ * (size_t)HDV_) / (256 * 8), 256, 0, stream>>>(Wo, wob);
  // 2) V -> V^T bf16
  k_vt<<<dim3(TK_ / 32, D_ / 32, B_), 256, 0, stream>>>(V, vt);
  // 3) Q/K projections (8192 wave-tiles / 8 waves per block)
  k_proj<<<1024, 256, 0, stream>>>(Q, K, Wq, bq, Wk, bk, ql, kl);
  // 4) flash attention: grid (TQ/16, B*H)
  k_attn<<<dim3(TQ_ / 16, B_ * H_), 256, 0, stream>>>(ql, kl, vt, mask, ao);
  // 5) output projection: 256*64 tiles / 8 waves per block
  k_oproj<<<(256 * 64) / 8, 256, 0, stream>>>(ao, wob, bo, out);
}
